// AdaptiveSpanAttention_53669911330827
// MI455X (gfx1250) — matrix operand order repaired and checked by reading, not compiled, not run on hardware
//
#include <hip/hip_runtime.h>
#include <hip/hip_bf16.h>
#include <math.h>

typedef __bf16 bf16;
typedef bf16  bf16x4  __attribute__((ext_vector_type(4)));
typedef bf16  bf16x8  __attribute__((ext_vector_type(8)));
typedef bf16  bf16x16 __attribute__((ext_vector_type(16)));
typedef float floatx8 __attribute__((ext_vector_type(8)));

#define S_LEN 1024
#define BATCH 64
#define EDIM  1024
#define NROWS (S_LEN * BATCH)   // 65536

__device__ __forceinline__ float gelu_exact(float x) {
    return 0.5f * x * (1.0f + erff(x * 0.70710678118654752f));
}
__device__ __forceinline__ float sigmoidf(float x) {
    return 1.0f / (1.0f + expf(-x));
}

// ---------------------------------------------------------------- ctx = mean_t encoded_x
__global__ __launch_bounds__(256) void k_ctx(const float* __restrict__ enc,
                                             float* __restrict__ ctx) {
    int idx = blockIdx.x * 256 + threadIdx.x;           // 65536 = B*E
    float acc = 0.f;
    for (int t = 0; t < 256; ++t) acc += enc[t * (BATCH * EDIM) + idx];
    ctx[idx] = acc * (1.0f / 256.0f);
}

// ---------------------------------------------------------------- span predictor MLP
__global__ __launch_bounds__(256) void k_span(const float* __restrict__ ctx,
                                              const float* __restrict__ w1, const float* __restrict__ b1,
                                              const float* __restrict__ w2, const float* __restrict__ b2,
                                              const float* __restrict__ w3, const float* __restrict__ b3,
                                              int* __restrict__ spans) {
    __shared__ float cx[1024];
    __shared__ float h1[512];
    __shared__ float h2[256];
    __shared__ float red[256];
    const int b = blockIdx.x, t = threadIdx.x;
    for (int i = t; i < 1024; i += 256) cx[i] = ctx[b * 1024 + i];
    __syncthreads();
    for (int c = t; c < 512; c += 256) {
        float acc = b1[c];
        for (int k = 0; k < 1024; ++k) acc += cx[k] * w1[k * 512 + c];
        h1[c] = gelu_exact(acc);
    }
    __syncthreads();
    {
        float acc = b2[t];
        for (int k = 0; k < 512; ++k) acc += h1[k] * w2[k * 256 + t];
        h2[t] = gelu_exact(acc);
    }
    __syncthreads();
    red[t] = h2[t] * w3[t];
    __syncthreads();
    for (int off = 128; off > 0; off >>= 1) {
        if (t < off) red[t] += red[t + off];
        __syncthreads();
    }
    if (t == 0) {
        float sig  = sigmoidf(red[0] + b3[0]);
        int   span = (int)(sig * 1016.0f + 8.0f);
        spans[b] = span > S_LEN ? S_LEN : span;
    }
}

// ---------------------------------------------------------------- ctx_bias = ctx @ rs_w1[E:] + rs_b1
__global__ __launch_bounds__(256) void k_bias(const float* __restrict__ ctx,
                                              const float* __restrict__ rs_w1,
                                              const float* __restrict__ rs_b1,
                                              float* __restrict__ bias) {
    const int b = blockIdx.x >> 2;
    const int n = ((blockIdx.x & 3) << 8) + threadIdx.x;
    const float* w = rs_w1 + (size_t)EDIM * EDIM;   // rows E..2E-1
    float acc = rs_b1[n];
    for (int k = 0; k < EDIM; ++k) acc += ctx[b * EDIM + k] * w[(size_t)k * EDIM + n];
    bias[b * EDIM + n] = acc;
}

// ---------------------------------------------------------------- pack weights to bf16 fragment layout
// Tile (kt, nt) of 32x16; per lane l (0..31), element e (0..15):
//   K_local = 16*(e>=8) + 8*(l>=16) + (e&7), N_local = l&15   (V_WMMA B-matrix 32x16 bf16 layout)
__global__ __launch_bounds__(256) void k_pack(const float* __restrict__ w,
                                              bf16* __restrict__ wp,
                                              int N /*cols*/, int total) {
    int gid = blockIdx.x * 256 + threadIdx.x;
    if (gid >= total) return;
    int e    = gid & 15;
    int l    = (gid >> 4) & 31;
    int tile = gid >> 9;
    int ntiles = N >> 4;
    int nt = tile % ntiles;
    int kt = tile / ntiles;
    int k  = kt * 32 + ((e >> 3) << 4) + ((l >> 4) << 3) + (e & 7);
    int n  = nt * 16 + (l & 15);
    wp[gid] = (bf16)w[(size_t)k * N + n];
}

// ---------------------------------------------------------------- fused relevance scorer
// Per block: 32 rows (2 M-tiles) of long_x. Each B fragment feeds two WMMAs, halving
// L2 weight traffic vs a 16-row strip. B fragments are loaded in batches of 4 so the
// compiler emits one load clause and overlapping waits instead of load->wait0->wmma.
__global__ __launch_bounds__(256) void k_fused(const float* __restrict__ long_x,
                                               const bf16* __restrict__ w1p,
                                               const bf16* __restrict__ w2p,
                                               const float* __restrict__ ctx_bias,
                                               const float* __restrict__ rs_b2,
                                               const float* __restrict__ rs_w3,
                                               const float* __restrict__ rs_b3,
                                               float* __restrict__ rel) {
    __shared__ bf16 smem[32 * 1024];                    // 64 KB, reused for X -> h -> r2
    const int tid  = threadIdx.x;
    const int lane = tid & 31;
    const int wave = tid >> 5;
    const int r0   = blockIdx.x * 32;
    const int mrow   = lane & 15;
    const int khalfA = (lane >> 4) * 8;

    // ---- load 32x1024 fp32 strip, convert to bf16 in LDS (strip is contiguous) ----
    {
        const float4* src4 = (const float4*)(long_x + (size_t)r0 * EDIM);
        for (int i = tid; i < 32 * 1024 / 4; i += 256) {
            float4 v = src4[i];
            bf16x4 p;
            p[0] = (bf16)v.x; p[1] = (bf16)v.y; p[2] = (bf16)v.z; p[3] = (bf16)v.w;
            *(bf16x4*)(&smem[i * 4]) = p;
        }
    }
    __syncthreads();

    // ---- GEMM1: wave owns cols [wave*128, wave*128+128) for both M-tiles ----
    floatx8 c1[2][8];
#pragma unroll
    for (int mt = 0; mt < 2; ++mt)
#pragma unroll
        for (int n = 0; n < 8; ++n) c1[mt][n] = (floatx8)0.0f;
    const int n0 = wave * 128;
    for (int kt = 0; kt < 32; ++kt) {
        bf16x16 a[2];
#pragma unroll
        for (int mt = 0; mt < 2; ++mt) {
            const bf16* ap = &smem[(mt * 16 + mrow) * 1024 + kt * 32 + khalfA];
            bf16x8 alo = *(const bf16x8*)ap;
            bf16x8 ahi = *(const bf16x8*)(ap + 16);
#pragma unroll
            for (int i = 0; i < 8; ++i) { a[mt][i] = alo[i]; a[mt][i + 8] = ahi[i]; }
        }
#pragma unroll
        for (int half = 0; half < 2; ++half) {
            bf16x16 bfrag[4];
#pragma unroll
            for (int j = 0; j < 4; ++j) {
                int nt = half * 4 + j;
                bfrag[j] = *(const bf16x16*)(w1p +
                    ((size_t)(kt * 64 + (n0 >> 4) + nt) * 512) + lane * 16);
            }
#pragma unroll
            for (int j = 0; j < 4; ++j) {
                int nt = half * 4 + j;
                c1[0][nt] = __builtin_amdgcn_wmma_f32_16x16x32_bf16(
                    false, a[0], false, bfrag[j], (short)0, c1[0][nt], false, false);
                c1[1][nt] = __builtin_amdgcn_wmma_f32_16x16x32_bf16(
                    false, a[1], false, bfrag[j], (short)0, c1[1][nt], false, false);
            }
        }
    }
    __syncthreads();   // all waves finished reading X strip

    // ---- bias + GELU -> h (bf16) back into smem ----
    const int bb = r0 & 63;   // batch offset of row 0 (no wrap inside a 32-row strip)
#pragma unroll
    for (int mt = 0; mt < 2; ++mt)
#pragma unroll
    for (int nt = 0; nt < 8; ++nt) {
        int col = n0 + nt * 16 + (lane & 15);
#pragma unroll
        for (int v = 0; v < 8; ++v) {
            int M = mt * 16 + v + ((lane >> 4) << 3);
            float x = c1[mt][nt][v] + ctx_bias[(bb + M) * EDIM + col];
            smem[M * 1024 + col] = (bf16)gelu_exact(x);
        }
    }
    __syncthreads();

    // ---- GEMM2: wave owns cols [wave*64, wave*64+64) of 512, both M-tiles ----
    floatx8 c2[2][4];
#pragma unroll
    for (int mt = 0; mt < 2; ++mt)
#pragma unroll
        for (int n = 0; n < 4; ++n) c2[mt][n] = (floatx8)0.0f;
    const int n0b = wave * 64;
    for (int kt = 0; kt < 32; ++kt) {
        bf16x16 a[2];
#pragma unroll
        for (int mt = 0; mt < 2; ++mt) {
            const bf16* ap = &smem[(mt * 16 + mrow) * 1024 + kt * 32 + khalfA];
            bf16x8 alo = *(const bf16x8*)ap;
            bf16x8 ahi = *(const bf16x8*)(ap + 16);
#pragma unroll
            for (int i = 0; i < 8; ++i) { a[mt][i] = alo[i]; a[mt][i + 8] = ahi[i]; }
        }
        bf16x16 bfrag[4];
#pragma unroll
        for (int nt = 0; nt < 4; ++nt) {
            bfrag[nt] = *(const bf16x16*)(w2p +
                ((size_t)(kt * 32 + (n0b >> 4) + nt) * 512) + lane * 16);
        }
#pragma unroll
        for (int nt = 0; nt < 4; ++nt) {
            c2[0][nt] = __builtin_amdgcn_wmma_f32_16x16x32_bf16(
                false, a[0], false, bfrag[nt], (short)0, c2[0][nt], false, false);
            c2[1][nt] = __builtin_amdgcn_wmma_f32_16x16x32_bf16(
                false, a[1], false, bfrag[nt], (short)0, c2[1][nt], false, false);
        }
    }
    __syncthreads();   // done reading h

    // ---- bias + GELU -> r2 (bf16) into smem[0 .. 32*512) ----
#pragma unroll
    for (int mt = 0; mt < 2; ++mt)
#pragma unroll
    for (int nt = 0; nt < 4; ++nt) {
        int col = n0b + nt * 16 + (lane & 15);
#pragma unroll
        for (int v = 0; v < 8; ++v) {
            int M = mt * 16 + v + ((lane >> 4) << 3);
            float x = c2[mt][nt][v] + rs_b2[col];
            smem[M * 512 + col] = (bf16)gelu_exact(x);
        }
    }
    __syncthreads();

    // ---- final dot with rs_w3 + sigmoid ----
    if (tid < 32) {
        float acc = rs_b3[0];
        for (int k = 0; k < 512; ++k) acc += (float)smem[tid * 512 + k] * rs_w3[k];
        rel[r0 + tid] = sigmoidf(acc);   // flat [S,B] index == row index
    }
}

// ---------------------------------------------------------------- finalize attn, spans, zero feats
__global__ __launch_bounds__(256) void k_attn(const float* __restrict__ rel,
                                              const int* __restrict__ spans,
                                              const float* __restrict__ head_scale,
                                              const float* __restrict__ td,
                                              const float* __restrict__ thr,
                                              const float* __restrict__ reg,
                                              float* __restrict__ out) {
    __shared__ float red[256];
    const int b = blockIdx.x, t = threadIdx.x;
    float hs = 0.f;
    for (int i = 0; i < 16; ++i) hs += head_scale[i];
    hs *= (1.0f / 16.0f);
    const int   span  = spans[b];
    const int   start = S_LEN - span;
    const float tdv = td[0], thv = thr[0], rg = reg[0];

    float fs[4];
    float partial = 0.f;
#pragma unroll
    for (int j = 0; j < 4; ++j) {
        int s = t + j * 256;
        float decay = powf(tdv, (float)(S_LEN - 1 - s));
        float f = rel[s * BATCH + b] * decay;
        f = (f > thv) ? f : 0.f;
        f *= hs;
        f = (s >= start) ? f : 0.f;
        fs[j] = f;
        partial += f;
    }
    red[t] = partial;
    __syncthreads();
    for (int off = 128; off > 0; off >>= 1) {
        if (t < off) red[t] += red[t + off];
        __syncthreads();
    }
    const float total = red[0];
    const float inv = 1.0f / (total + rg + 1e-8f);
    const float uni = 1.0f / (float)span;
#pragma unroll
    for (int j = 0; j < 4; ++j) {
        int s = t + j * 256;
        float a = (total > 0.f) ? fs[j] * inv : ((s >= start) ? uni : 0.f);
        out[s * BATCH + b] = a;
    }
    // zero feats region for the atomic accumulation pass
#pragma unroll
    for (int j = 0; j < 4; ++j) out[NROWS + b * EDIM + t + j * 256] = 0.f;
    if (t == 0) ((int*)out)[2 * NROWS + b] = span;
}

// ---------------------------------------------------------------- feats[b,e] = sum_s attn[s,b]*long_x[s,b,e]
__global__ __launch_bounds__(256) void k_feats(const float* __restrict__ long_x,
                                               const float* __restrict__ attn,
                                               float* __restrict__ out) {
    const int bx   = blockIdx.x;            // 1024 blocks: b(64) x echunk(4) x sseg(4)
    const int b    = bx & 63;
    const int ech  = (bx >> 6) & 3;
    const int sseg = bx >> 8;
    const int e    = ech * 256 + threadIdx.x;
    float acc = 0.f;
    for (int s = sseg * 256; s < sseg * 256 + 256; ++s) {
        float a = attn[s * BATCH + b];
        acc += a * long_x[(size_t)(s * BATCH + b) * EDIM + e];
    }
    atomicAdd(&out[NROWS + b * EDIM + e], acc);
}

// ----------------------------------------------------------------
extern "C" void kernel_launch(void* const* d_in, const int* in_sizes, int n_in,
                              void* d_out, int out_size, void* d_ws, size_t ws_size,
                              hipStream_t stream) {
    const float* long_x    = (const float*)d_in[0];
    const float* encoded_x = (const float*)d_in[1];
    const float* sp_w1 = (const float*)d_in[2];
    const float* sp_b1 = (const float*)d_in[3];
    const float* sp_w2 = (const float*)d_in[4];
    const float* sp_b2 = (const float*)d_in[5];
    const float* sp_w3 = (const float*)d_in[6];
    const float* sp_b3 = (const float*)d_in[7];
    const float* rs_w1 = (const float*)d_in[8];
    const float* rs_b1 = (const float*)d_in[9];
    const float* rs_w2 = (const float*)d_in[10];
    const float* rs_b2 = (const float*)d_in[11];
    const float* rs_w3 = (const float*)d_in[12];
    const float* rs_b3 = (const float*)d_in[13];
    const float* head_scale = (const float*)d_in[14];
    const float* td  = (const float*)d_in[15];
    const float* thr = (const float*)d_in[16];
    const float* reg = (const float*)d_in[17];

    char* ws = (char*)d_ws;
    float* ctx   = (float*)(ws);                  // 256 KB
    float* bias  = (float*)(ws + 0x40000);        // 256 KB
    float* rel   = (float*)(ws + 0x80000);        // 256 KB
    int*   spans = (int*)  (ws + 0xC0000);        // tiny
    bf16*  w1p   = (bf16*) (ws + 0x100000);       // 2 MB packed bf16
    bf16*  w2p   = (bf16*) (ws + 0x300000);       // 1 MB packed bf16
    float* out   = (float*)d_out;

    k_ctx  <<<256, 256, 0, stream>>>(encoded_x, ctx);
    k_span <<<64,  256, 0, stream>>>(ctx, sp_w1, sp_b1, sp_w2, sp_b2, sp_w3, sp_b3, spans);
    k_bias <<<256, 256, 0, stream>>>(ctx, rs_w1, rs_b1, bias);
    k_pack <<<4096, 256, 0, stream>>>(rs_w1, w1p, 1024, 1024 * 1024);
    k_pack <<<2048, 256, 0, stream>>>(rs_w2, w2p, 512, 1024 * 512);
    k_fused<<<2048, 256, 0, stream>>>(long_x, w1p, w2p, bias, rs_b2, rs_w3, rs_b3, rel);
    k_attn <<<64,  256, 0, stream>>>(rel, spans, head_scale, td, thr, reg, out);
    k_feats<<<1024, 256, 0, stream>>>(long_x, out, out);
}
